// CODABlocks_9483287789756
// MI455X (gfx1250) — matrix operand order repaired and checked by reading, not compiled, not run on hardware
//
#include <hip/hip_runtime.h>
#include <math.h>

// ---------------------------------------------------------------------------
// CODA block for MI455X (gfx1250, wave32).  All FFTs are truncated-mode DFTs
// expressed as batched GEMMs running on v_wmma_f32_16x16x32_f16.  The two
// attention GEMMs stage their tiles through the Tensor Data Mover.
// ---------------------------------------------------------------------------

typedef __attribute__((ext_vector_type(16))) _Float16 v16h;
typedef __attribute__((ext_vector_type(8)))  _Float16 v8h;
typedef __attribute__((ext_vector_type(4)))  _Float16 v4h;
typedef __attribute__((ext_vector_type(8)))  float    v8f;
typedef __attribute__((ext_vector_type(4)))  unsigned int v4u;
typedef __attribute__((ext_vector_type(8)))  int      v8i;
typedef __attribute__((ext_vector_type(4)))  int      v4i;

#define GT_M 32
#define GT_N 64
#define GT_K 32
#define GT_PAD 8   // f16 row padding -> 80-byte rows (16B aligned)
#define CDIV(a,b) (((a)+(b)-1)/(b))

// --------------------- fragment build + wmma (shared) ----------------------
__device__ __forceinline__ v8f wmma_tile(const _Float16* sA, const _Float16* sBt,
                                         int lane, int wm, int wn, v8f acc)
{
  const int arow = wm * 16 + (lane & 15);
  const int akb  = (lane >= 16) ? 8 : 0;
  const int ldr  = GT_K + GT_PAD;
  // A fragment: lanes 0-15 -> K {0..7,16..23}; lanes 16-31 -> K {8..15,24..31}
  v8h a0 = *(const v8h*)&sA[arow * ldr + akb];
  v8h a1 = *(const v8h*)&sA[arow * ldr + akb + 16];
  v16h af = __builtin_shufflevector(a0, a1, 0,1,2,3,4,5,6,7,8,9,10,11,12,13,14,15);
  // B fragment: lanes 0-15 -> K {0..7,16..23}; lanes 16-31 -> K {8..15,24..31} (sBt is [n][k])
  const int bcol = wn * 16 + (lane & 15);
  const int bkb  = (lane >= 16) ? 8 : 0;
  v8h b0 = *(const v8h*)&sBt[bcol * ldr + bkb];
  v8h b1 = *(const v8h*)&sBt[bcol * ldr + bkb + 16];
  v16h bf = __builtin_shufflevector(b0, b1, 0,1,2,3,4,5,6,7,8,9,10,11,12,13,14,15);
  return __builtin_amdgcn_wmma_f32_16x16x32_f16(false, af, false, bf,
                                                (short)0, acc, false, false);
}

// ------------------------- generic batched WMMA GEMM -----------------------
// C[z] = alpha * A[z] @ B[z].  Per-batch offset = (z/div)*s1 + (z%div)*s2.
// transB: B element (k,n) read as B[n*ldb + k].  Branch-free guarded loads.
__global__ __launch_bounds__(256) void gemm_wmma(
    const float* __restrict__ A, const float* __restrict__ B, float* __restrict__ C,
    int M, int N, int K, int lda, int ldb, int ldc, int transB, float alpha,
    int Adiv, long long As1, long long As2,
    int Bdiv, long long Bs1, long long Bs2,
    int Cdiv, long long Cs1, long long Cs2)
{
  const int z = blockIdx.z;
  const float* Ab = A + (long long)(z / Adiv) * As1 + (long long)(z % Adiv) * As2;
  const float* Bb = B + (long long)(z / Bdiv) * Bs1 + (long long)(z % Bdiv) * Bs2;
  float*       Cb = C + (long long)(z / Cdiv) * Cs1 + (long long)(z % Cdiv) * Cs2;

  __shared__ __align__(16) _Float16 sA [GT_M * (GT_K + GT_PAD)];
  __shared__ __align__(16) _Float16 sBt[GT_N * (GT_K + GT_PAD)];

  const int tid  = threadIdx.x;
  const int lane = tid & 31;
  const int wave = tid >> 5;
  const int wm = wave >> 2;
  const int wn = wave & 3;
  const int m0 = blockIdx.y * GT_M;
  const int n0 = blockIdx.x * GT_N;
  const int ldr = GT_K + GT_PAD;

  v8f acc = {0.f,0.f,0.f,0.f,0.f,0.f,0.f,0.f};

  for (int k0 = 0; k0 < K; k0 += GT_K) {
    // ---- stage A tile (32x32): thread -> 4 consecutive k, branch-free
    {
      int r = tid >> 3;
      int c = (tid & 7) << 2;
      int gm = m0 + r;
      bool mok = gm < M;
      long long base = (long long)gm * lda + k0 + c;
      _Float16 h[4];
#pragma unroll
      for (int j = 0; j < 4; ++j) {
        bool ok = mok && (k0 + c + j) < K;
        float v = Ab[ok ? (base + j) : 0];
        h[j] = (_Float16)(ok ? v : 0.f);
      }
      *(v4h*)&sA[r * ldr + c] = (v4h){h[0], h[1], h[2], h[3]};
    }
    // ---- stage B tile transposed (sBt[n][k]): thread -> 4 consecutive k
#pragma unroll
    for (int pass = 0; pass < 2; ++pass) {
      int n = tid & 63;
      int kk = (((tid >> 6) + pass * 4)) << 2;  // 0,4,8,12 / 16,20,24,28
      int gn = n0 + n;
      bool nok = gn < N;
      _Float16 h[4];
#pragma unroll
      for (int j = 0; j < 4; ++j) {
        int gk = k0 + kk + j;
        bool ok = nok && gk < K;
        long long id = transB ? ((long long)gn * ldb + gk)
                              : ((long long)gk * ldb + gn);
        float v = Bb[ok ? id : 0];
        h[j] = (_Float16)(ok ? v : 0.f);
      }
      *(v4h*)&sBt[n * ldr + kk] = (v4h){h[0], h[1], h[2], h[3]};
    }
    __syncthreads();

    if (k0 + GT_K < K)  // gfx1250 global_prefetch_b8 hint for next A tile
      __builtin_prefetch(Ab + (long long)(m0 + lane) * lda + k0 + GT_K, 0, 1);

    acc = wmma_tile(sA, sBt, lane, wm, wn, acc);
    __syncthreads();
  }

  const int n  = n0 + wn * 16 + (lane & 15);
  const int mb = m0 + wm * 16 + ((lane >= 16) ? 8 : 0);
  if (n < N) {
#pragma unroll
    for (int r = 0; r < 8; ++r) {
      int m = mb + r;
      if (m < M) Cb[(long long)m * ldc + n] = acc[r] * alpha;
    }
  }
}

// --------------------- TDM-staged WMMA GEMM (full tiles) -------------------
// Requires M%32==0, N%64==0, K%32==0.  A and B tiles are DMA'd to LDS by the
// Tensor Data Mover (2D D#), then converted f32->f16 for the matrix unit.
__device__ __forceinline__ void tdm_load_2d(unsigned lds_addr, const float* gptr,
                                            int tile_w, int tile_h,
                                            long long row_stride_elems)
{
  unsigned long long ga = (unsigned long long)(size_t)gptr;
  v4u g0;
  g0.x = 1u;                                   // count=1 (valid descriptor)
  g0.y = lds_addr;                             // LDS byte address
  g0.z = (unsigned)(ga & 0xFFFFFFFFull);       // global_addr[31:0]
  g0.w = (unsigned)((ga >> 32) & 0x1FFFFFFull) // global_addr[56:32]
       | 0x80000000u;                          // type=2 ("image")
  const unsigned td0 = 1u << 20, td1 = 1u << 20;  // tiles provably in-bounds
  unsigned long long st0 = (unsigned long long)row_stride_elems;
  v8i g1;
  g1[0] = 0x20000;                                             // data_size=4B
  g1[1] = (int)((td0 & 0xFFFFu) << 16);                        // tensor_dim0 lo
  g1[2] = (int)(((td0 >> 16) & 0xFFFFu) | ((td1 & 0xFFFFu) << 16));
  g1[3] = (int)(((td1 >> 16) & 0xFFFFu) | ((unsigned)tile_w << 16)); // tile_dim0
  g1[4] = (int)((unsigned)tile_h & 0xFFFFu);                   // tile_dim1
  g1[5] = (int)(st0 & 0xFFFFFFFFull);                          // dim0 stride lo
  g1[6] = (int)((st0 >> 32) & 0xFFFFull);                      // dim0 stride hi
  g1[7] = 0;
  v4i z4 = {0, 0, 0, 0};
  v8i z8 = {0, 0, 0, 0, 0, 0, 0, 0};
  __builtin_amdgcn_tensor_load_to_lds(g0, g1, z4, z4, z8, 0);
}

__global__ __launch_bounds__(256) void gemm_wmma_tdm(
    const float* __restrict__ A, const float* __restrict__ B, float* __restrict__ C,
    int M, int N, int K, int lda, int ldb, int ldc, int transB, float alpha,
    int Adiv, long long As1, long long As2,
    int Bdiv, long long Bs1, long long Bs2,
    int Cdiv, long long Cs1, long long Cs2)
{
  const int z = blockIdx.z;
  const float* Ab = A + (long long)(z / Adiv) * As1 + (long long)(z % Adiv) * As2;
  const float* Bb = B + (long long)(z / Bdiv) * Bs1 + (long long)(z % Bdiv) * Bs2;
  float*       Cb = C + (long long)(z / Cdiv) * Cs1 + (long long)(z % Cdiv) * Cs2;

  __shared__ __align__(16) float    stg[GT_M * GT_K + GT_K * GT_N]; // f32 staging
  __shared__ __align__(16) _Float16 sA [GT_M * (GT_K + GT_PAD)];
  __shared__ __align__(16) _Float16 sBt[GT_N * (GT_K + GT_PAD)];

  const int tid  = threadIdx.x;
  const int lane = tid & 31;
  const int wave = tid >> 5;
  const int wm = wave >> 2;
  const int wn = wave & 3;
  const int m0 = blockIdx.y * GT_M;
  const int n0 = blockIdx.x * GT_N;
  const int ldr = GT_K + GT_PAD;

  float* stgA = stg;                    // [32][32]
  float* stgB = stg + GT_M * GT_K;      // !transB: [32][64] ; transB: [64][32]
  const unsigned ldsA = (unsigned)(size_t)(void*)stgA;
  const unsigned ldsB = (unsigned)(size_t)(void*)stgB;

  v8f acc = {0.f,0.f,0.f,0.f,0.f,0.f,0.f,0.f};

  for (int k0 = 0; k0 < K; k0 += GT_K) {
    if (tid < 32) {  // wave 0 drives the Tensor Data Mover
      tdm_load_2d(ldsA, Ab + (long long)m0 * lda + k0, GT_K, GT_M, lda);
      if (transB)
        tdm_load_2d(ldsB, Bb + (long long)n0 * ldb + k0, GT_K, GT_N, ldb);
      else
        tdm_load_2d(ldsB, Bb + (long long)k0 * ldb + n0, GT_N, GT_K, ldb);
      __builtin_amdgcn_s_wait_tensorcnt(0);
    }
    __syncthreads();                    // staged f32 tiles visible to all waves

    // convert A: thread -> 4 consecutive k of one row
    {
      int r = tid >> 3;
      int c = (tid & 7) << 2;
      v4h h;
#pragma unroll
      for (int j = 0; j < 4; ++j) h[j] = (_Float16)stgA[r * GT_K + c + j];
      *(v4h*)&sA[r * ldr + c] = h;
    }
    // convert B into transposed f16 tile sBt[n][k]: thread -> 8 k of one n
    {
      int n  = tid & 63;
      int kb = (tid >> 6) << 3;
      v8h h;
#pragma unroll
      for (int j = 0; j < 8; ++j) {
        float v = transB ? stgB[n * GT_K + kb + j] : stgB[(kb + j) * GT_N + n];
        h[j] = (_Float16)v;
      }
      *(v8h*)&sBt[n * ldr + kb] = h;
    }
    __syncthreads();

    acc = wmma_tile(sA, sBt, lane, wm, wn, acc);
    __syncthreads();                    // staging reusable next iteration
  }

  const int n  = n0 + wn * 16 + (lane & 15);
  const int mb = m0 + wm * 16 + ((lane >= 16) ? 8 : 0);
#pragma unroll
  for (int r = 0; r < 8; ++r)
    Cb[(long long)(mb + r) * ldc + n] = acc[r] * alpha;
}

// ------------------------------ DFT basis fills ----------------------------
__global__ void fill_fwd_width(float* wc, float* wsn, int W, int m2h) {
  int i = blockIdx.x * 256 + threadIdx.x;
  if (i >= W * m2h) return;
  int w = i / m2h, j = i % m2h;
  float th = 6.28318530717958647692f * (float)(((long long)w * j) % W) / (float)W;
  wc[i]  =  cosf(th) / (float)W;
  wsn[i] = -sinf(th) / (float)W;
}

__global__ void fill_fwd_height(float* out, int H, int hh) {
  int cols = 2 * H, rows = 4 * hh;
  int i = blockIdx.x * 256 + threadIdx.x;
  if (i >= rows * cols) return;
  int row = i / cols, k = i % cols;
  int rho = (row < 2 * hh) ? row : row - 2 * hh;
  int ky  = (rho < hh) ? rho : H - 2 * hh + rho;
  int r   = (k < H) ? k : k - H;
  float th = 6.28318530717958647692f * (float)(((long long)ky * r) % H) / (float)H;
  float c = cosf(th) / (float)H, s = sinf(th) / (float)H;
  out[i] = (row < 2 * hh) ? ((k < H) ? c : s) : ((k < H) ? -s : c);
}

__global__ void fill_invA(float* oRe, float* oIm, int Ho, int hh) {
  int cols = 4 * hh;
  int i = blockIdx.x * 256 + threadIdx.x;
  if (i >= Ho * cols) return;
  int h = i / cols, k = i % cols;
  int rho = (k < 2 * hh) ? k : k - 2 * hh;
  int ky  = (rho < hh) ? rho : Ho - 2 * hh + rho;
  float th = 6.28318530717958647692f * (float)(((long long)ky * h) % Ho) / (float)Ho;
  float c = cosf(th), s = sinf(th);
  oRe[i] = (k < 2 * hh) ? c : -s;
  oIm[i] = (k < 2 * hh) ? s :  c;
}

__global__ void fill_invB(float* out, int Wo, int m2h) {
  int i = blockIdx.x * 256 + threadIdx.x;
  if (i >= 2 * m2h * Wo) return;
  int row = i / Wo, w = i % Wo;
  int j = (row < m2h) ? row : row - m2h;
  float cj = (j == 0 || 2 * j == Wo) ? 1.f : 2.f;
  float th = 6.28318530717958647692f * (float)(((long long)j * w) % Wo) / (float)Wo;
  out[i] = (row < m2h) ? cj * cosf(th) : -cj * sinf(th);
}

// ------------------------------ elementwise --------------------------------
__global__ void inorm_kernel(const float* __restrict__ x, const float* __restrict__ y,
                             const float* __restrict__ zadd, float* __restrict__ out,
                             int HW, const float* g, const float* b)
{
  int img = blockIdx.x, tid = threadIdx.x;
  const float* xi = x + (long long)img * HW;
  const float* yi = y ? y + (long long)img * HW : nullptr;
  __shared__ float s1[256], s2[256];
  float sum = 0.f, sq = 0.f;
  for (int i = tid; i < HW; i += 256) {
    float v = xi[i] + (yi ? yi[i] : 0.f);
    sum += v; sq += v * v;
  }
  s1[tid] = sum; s2[tid] = sq; __syncthreads();
  for (int o = 128; o > 0; o >>= 1) {
    if (tid < o) { s1[tid] += s1[tid + o]; s2[tid] += s2[tid + o]; }
    __syncthreads();
  }
  float mu  = s1[0] / (float)HW;
  float var = s2[0] / (float)HW - mu * mu;
  float inv = rsqrtf(var + 1e-5f);
  float gg = g ? g[0] : 1.f, bb = b ? b[0] : 0.f;
  const float* zi = zadd ? zadd + (long long)img * HW : nullptr;
  float* oi = out + (long long)img * HW;
  for (int i = tid; i < HW; i += 256) {
    float v = xi[i] + (yi ? yi[i] : 0.f);
    float r = (v - mu) * inv * gg + bb;
    if (zi) r += zi[i];
    oi[i] = r;
  }
}

__global__ void mode_mul_expand(const float* __restrict__ in, const float* __restrict__ w,
                                float* __restrict__ out, int imgs, int heads, int rows, int m2h)
{
  long long total = (long long)imgs * heads * rows * m2h;
  long long i = (long long)blockIdx.x * 256 + threadIdx.x;
  if (i >= total) return;
  int kx = (int)(i % m2h); long long t = i / m2h;
  int rho = (int)(t % rows); t /= rows;
  int h = (int)(t % heads); int img = (int)(t / heads);
  long long per = (long long)2 * rows * m2h;
  const float* ip = in + (long long)img * per;
  float re = ip[(long long)rho * m2h + kx];
  float im = ip[(long long)(rows + rho) * m2h + kx];
  long long wi = (((long long)h * rows + rho) * m2h + kx) * 2;
  float wr = w[wi], wim = w[wi + 1];
  float* op = out + ((long long)img * heads + h) * per;
  op[(long long)rho * m2h + kx]          = re * wr - im * wim;
  op[(long long)(rows + rho) * m2h + kx] = re * wim + im * wr;
}

__global__ void mode_mul_reduce(const float* __restrict__ in, const float* __restrict__ w,
                                float* __restrict__ out, int imgs, int heads, int rows, int m2h)
{
  long long total = (long long)imgs * rows * m2h;
  long long i = (long long)blockIdx.x * 256 + threadIdx.x;
  if (i >= total) return;
  int kx = (int)(i % m2h); long long t = i / m2h;
  int rho = (int)(t % rows); int img = (int)(t / rows);
  long long per = (long long)2 * rows * m2h;
  float are = 0.f, aim = 0.f;
  for (int h = 0; h < heads; ++h) {
    const float* ip = in + ((long long)img * heads + h) * per;
    float re = ip[(long long)rho * m2h + kx];
    float im = ip[(long long)(rows + rho) * m2h + kx];
    long long wi = (((long long)h * rows + rho) * m2h + kx) * 2;
    float wr = w[wi], wim = w[wi + 1];
    are += re * wr - im * wim;
    aim += re * wim + im * wr;
  }
  float* op = out + (long long)img * per;
  op[(long long)rho * m2h + kx]          = are;
  op[(long long)(rows + rho) * m2h + kx] = aim;
}

__global__ void axpb_add(float* __restrict__ dst, const float* __restrict__ src,
                         const float* __restrict__ w, const float* __restrict__ b,
                         int imgs, int heads, int per)
{
  long long total = (long long)imgs * heads * per;
  long long i = (long long)blockIdx.x * 256 + threadIdx.x;
  if (i >= total) return;
  int p = (int)(i % per); long long t = i / per;
  int h = (int)(t % heads); int img = (int)(t / heads);
  dst[((long long)img * heads + h) * per + p] += w[h] * src[(long long)img * per + p] + b[h];
}

__global__ void proj_skip_add(float* __restrict__ dst, const float* __restrict__ src,
                              const float* __restrict__ w, const float* __restrict__ b,
                              int imgs, int heads, int per)
{
  long long total = (long long)imgs * per;
  long long i = (long long)blockIdx.x * 256 + threadIdx.x;
  if (i >= total) return;
  int p = (int)(i % per); int img = (int)(i / per);
  float acc = b[0];
  for (int h = 0; h < heads; ++h)
    acc += w[h] * src[((long long)img * heads + h) * per + p];
  dst[i] += acc;
}

__global__ void mixer_combine(float* __restrict__ out, const float* __restrict__ spec,
                              const float* __restrict__ base, const float* __restrict__ w,
                              const float* __restrict__ b, long long n, int do_gelu)
{
  long long i = (long long)blockIdx.x * 256 + threadIdx.x;
  if (i >= n) return;
  float v = spec[i] + w[0] * base[i] + b[0];
  if (do_gelu) v = 0.5f * v * (1.f + erff(v * 0.70710678118654752f));
  out[i] = v;
}

__global__ void softmax64(float* __restrict__ s) {
  int r = blockIdx.x, t = threadIdx.x;
  float* p = s + (long long)r * 64;
  __shared__ float sm[64];
  float v = p[t]; sm[t] = v; __syncthreads();
  for (int o = 32; o > 0; o >>= 1) { if (t < o) sm[t] = fmaxf(sm[t], sm[t + o]); __syncthreads(); }
  float mx = sm[0]; __syncthreads();
  float e = expf(v - mx); sm[t] = e; __syncthreads();
  for (int o = 32; o > 0; o >>= 1) { if (t < o) sm[t] += sm[t + o]; __syncthreads(); }
  p[t] = e / sm[0];
}

// ------------------------------ host driver --------------------------------
static inline void launch_gemm(hipStream_t st,
    const float* A, const float* B, float* C,
    int M, int N, int K, int lda, int ldb, int ldc,
    int batch, int transB, float alpha,
    int Adiv, long long As1, long long As2,
    int Bdiv, long long Bs1, long long Bs2,
    int Cdiv, long long Cs1, long long Cs2)
{
  dim3 grid(CDIV(N, GT_N), CDIV(M, GT_M), batch);
  gemm_wmma<<<grid, 256, 0, st>>>(A, B, C, M, N, K, lda, ldb, ldc, transB, alpha,
                                  Adiv, As1, As2, Bdiv, Bs1, Bs2, Cdiv, Cs1, Cs2);
}

static inline void launch_gemm_tdm(hipStream_t st,
    const float* A, const float* B, float* C,
    int M, int N, int K, int lda, int ldb, int ldc,
    int batch, int transB, float alpha,
    int Adiv, long long As1, long long As2,
    int Bdiv, long long Bs1, long long Bs2,
    int Cdiv, long long Cs1, long long Cs2)
{
  dim3 grid(N / GT_N, M / GT_M, batch);   // full tiles only
  gemm_wmma_tdm<<<grid, 256, 0, st>>>(A, B, C, M, N, K, lda, ldb, ldc, transB, alpha,
                                      Adiv, As1, As2, Bdiv, Bs1, Bs2, Cdiv, Cs1, Cs2);
}

// workspace layout (floats)
static constexpr long long O_WC9   = 0;
static constexpr long long O_WS9   = O_WC9   + 128LL*9;
static constexpr long long O_WC17  = O_WS9   + 128LL*9;
static constexpr long long O_WS17  = O_WC17  + 128LL*17;
static constexpr long long O_WC33  = O_WS17  + 128LL*17;
static constexpr long long O_WS33  = O_WC33  + 128LL*33;
static constexpr long long O_FH8   = O_WS33  + 128LL*33;
static constexpr long long O_FH16  = O_FH8   + 32LL*256;
static constexpr long long O_FH32  = O_FH16  + 64LL*256;
static constexpr long long O_IAKQ_RE = O_FH32 + 128LL*256;
static constexpr long long O_IAKQ_IM = O_IAKQ_RE + 64LL*32;
static constexpr long long O_IAV_RE  = O_IAKQ_IM + 64LL*32;
static constexpr long long O_IAV_IM  = O_IAV_RE  + 128LL*32;
static constexpr long long O_IAM_RE  = O_IAV_IM  + 128LL*32;
static constexpr long long O_IAM_IM  = O_IAM_RE  + 128LL*64;
static constexpr long long O_IARS_RE = O_IAM_IM  + 128LL*64;
static constexpr long long O_IARS_IM = O_IARS_RE + 64LL*128;
static constexpr long long O_WCSKQ = O_IARS_IM + 64LL*128;
static constexpr long long O_WCSV  = O_WCSKQ + 18LL*64;
static constexpr long long O_WCSM  = O_WCSV  + 18LL*128;
static constexpr long long O_WCSRS = O_WCSM  + 34LL*128;
static constexpr long long O_TN   = O_WCSRS + 66LL*64;
static constexpr long long O_SA   = O_TN   + 4194304LL;
static constexpr long long O_MOD  = O_SA   + 8912896LL;
static constexpr long long O_SMOD = O_MOD  + 2228224LL;
static constexpr long long O_MKQV = O_SMOD + 278528LL;
static constexpr long long O_TP   = O_MKQV + 73728LL;
static constexpr long long O_K    = O_TP   + 4718592LL;
static constexpr long long O_Q    = O_K    + 8388608LL;
static constexpr long long O_V    = O_Q    + 8388608LL;
static constexpr long long O_AO   = O_V    + 33554432LL;
static constexpr long long O_R    = O_AO   + 33554432LL;
static constexpr long long O_SC   = O_R    + 1048576LL;
static constexpr long long O_T1   = O_SC   + 131072LL;
static constexpr long long O_A    = O_T1   + 4194304LL;
static constexpr long long O_AN   = O_A    + 4194304LL;
static constexpr long long O_T2   = O_AN   + 4194304LL;
static constexpr long long O_H    = O_T2   + 4194304LL;

extern "C" void kernel_launch(void* const* d_in, const int* in_sizes, int n_in,
                              void* d_out, int out_size, void* d_ws, size_t ws_size,
                              hipStream_t stream)
{
  (void)in_sizes; (void)n_in; (void)out_size; (void)ws_size;
  const float* xin    = (const float*)d_in[0];
  const float* k_spec = (const float*)d_in[1];
  const float* k_sw   = (const float*)d_in[2];
  const float* k_sb   = (const float*)d_in[3];
  const float* q_spec = (const float*)d_in[4];
  const float* q_sw   = (const float*)d_in[5];
  const float* q_sb   = (const float*)d_in[6];
  const float* v_spec = (const float*)d_in[7];
  const float* v_sw   = (const float*)d_in[8];
  const float* v_sb   = (const float*)d_in[9];
  const float* p_spec = (const float*)d_in[10];
  const float* p_sw   = (const float*)d_in[11];
  const float* p_sb   = (const float*)d_in[12];
  const float* m1_spec= (const float*)d_in[13];
  const float* m1_sw  = (const float*)d_in[14];
  const float* m1_sb  = (const float*)d_in[15];
  const float* m2_spec= (const float*)d_in[16];
  const float* m2_sw  = (const float*)d_in[17];
  const float* m2_sb  = (const float*)d_in[18];
  const float* n1_g = (const float*)d_in[19]; const float* n1_b = (const float*)d_in[20];
  const float* an_g = (const float*)d_in[21]; const float* an_b = (const float*)d_in[22];
  const float* n2_g = (const float*)d_in[23]; const float* n2_b = (const float*)d_in[24];
  const float* mo_g = (const float*)d_in[25]; const float* mo_b = (const float*)d_in[26];
  float* out = (float*)d_out;
  float* ws  = (float*)d_ws;
  hipStream_t st = stream;

  // ---- DFT bases
  fill_fwd_width<<<CDIV(128*9,256),256,0,st>>>(ws+O_WC9,  ws+O_WS9,  128, 9);
  fill_fwd_width<<<CDIV(128*17,256),256,0,st>>>(ws+O_WC17, ws+O_WS17, 128, 17);
  fill_fwd_width<<<CDIV(128*33,256),256,0,st>>>(ws+O_WC33, ws+O_WS33, 128, 33);
  fill_fwd_height<<<CDIV(32*256,256),256,0,st>>>(ws+O_FH8,  128, 8);
  fill_fwd_height<<<CDIV(64*256,256),256,0,st>>>(ws+O_FH16, 128, 16);
  fill_fwd_height<<<CDIV(128*256,256),256,0,st>>>(ws+O_FH32, 128, 32);
  fill_invA<<<CDIV(64*32,256),256,0,st>>>(ws+O_IAKQ_RE, ws+O_IAKQ_IM, 64, 8);
  fill_invA<<<CDIV(128*32,256),256,0,st>>>(ws+O_IAV_RE,  ws+O_IAV_IM, 128, 8);
  fill_invA<<<CDIV(128*64,256),256,0,st>>>(ws+O_IAM_RE,  ws+O_IAM_IM, 128, 16);
  fill_invA<<<CDIV(64*128,256),256,0,st>>>(ws+O_IARS_RE, ws+O_IARS_IM, 64, 32);
  fill_invB<<<CDIV(18*64,256),256,0,st>>>(ws+O_WCSKQ, 64, 9);
  fill_invB<<<CDIV(18*128,256),256,0,st>>>(ws+O_WCSV, 128, 9);
  fill_invB<<<CDIV(34*128,256),256,0,st>>>(ws+O_WCSM, 128, 17);
  fill_invB<<<CDIV(66*64,256),256,0,st>>>(ws+O_WCSRS, 64, 33);

  // ---- norm1: tn = inorm(x)
  inorm_kernel<<<256,256,0,st>>>(xin, nullptr, nullptr, ws+O_TN, 16384, n1_g, n1_b);

  // ---- R = resample(tn, 128 -> 64)
  launch_gemm(st, ws+O_TN, ws+O_WC33, ws+O_SA,      128,33,128, 128,33,33, 256,0,1.f, 1,16384,0, 1,0,0, 1,8448,0);
  launch_gemm(st, ws+O_TN, ws+O_WS33, ws+O_SA+4224, 128,33,128, 128,33,33, 256,0,1.f, 1,16384,0, 1,0,0, 1,8448,0);
  launch_gemm(st, ws+O_FH32, ws+O_SA, ws+O_MOD,     128,33,256, 256,33,33, 256,0,1.f, 1,0,0, 1,8448,0, 1,4224,0);
  launch_gemm(st, ws+O_IARS_RE, ws+O_MOD, ws+O_TP,    64,33,128, 128,33,66, 256,0,1.f, 1,0,0, 1,4224,0, 1,4224,0);
  launch_gemm(st, ws+O_IARS_IM, ws+O_MOD, ws+O_TP+33, 64,33,128, 128,33,66, 256,0,1.f, 1,0,0, 1,4224,0, 1,4224,0);
  launch_gemm(st, ws+O_TP, ws+O_WCSRS, ws+O_R,        64,64,66,  66,64,64,  256,0,1.f, 1,4224,0, 1,0,0, 1,4096,0);

  // ---- forward truncated DFT of tn (shared by K/Q/V): modes 16x9
  launch_gemm(st, ws+O_TN, ws+O_WC9, ws+O_SA,      128,9,128, 128,9,9, 256,0,1.f, 1,16384,0, 1,0,0, 1,2304,0);
  launch_gemm(st, ws+O_TN, ws+O_WS9, ws+O_SA+1152, 128,9,128, 128,9,9, 256,0,1.f, 1,16384,0, 1,0,0, 1,2304,0);
  launch_gemm(st, ws+O_FH8, ws+O_SA, ws+O_MKQV,    32,9,256,  256,9,9, 256,0,1.f, 1,0,0, 1,2304,0, 1,288,0);

  // ---- K
  mode_mul_expand<<<CDIV(256*8*16*9,256),256,0,st>>>(ws+O_MKQV, k_spec, ws+O_MOD, 256, 8, 16, 9);
  launch_gemm(st, ws+O_IAKQ_RE, ws+O_MOD, ws+O_TP,   64,9,32, 32,9,18, 2048,0,1.f, 1,0,0, 1,288,0, 1,1152,0);
  launch_gemm(st, ws+O_IAKQ_IM, ws+O_MOD, ws+O_TP+9, 64,9,32, 32,9,18, 2048,0,1.f, 1,0,0, 1,288,0, 1,1152,0);
  launch_gemm(st, ws+O_TP, ws+O_WCSKQ, ws+O_K,       64,64,18, 18,64,64, 2048,0,1.f, 1,1152,0, 1,0,0, 1,4096,0);
  axpb_add<<<CDIV(2048*4096,256),256,0,st>>>(ws+O_K, ws+O_R, k_sw, k_sb, 256, 8, 4096);

  // ---- Q
  mode_mul_expand<<<CDIV(256*8*16*9,256),256,0,st>>>(ws+O_MKQV, q_spec, ws+O_MOD, 256, 8, 16, 9);
  launch_gemm(st, ws+O_IAKQ_RE, ws+O_MOD, ws+O_TP,   64,9,32, 32,9,18, 2048,0,1.f, 1,0,0, 1,288,0, 1,1152,0);
  launch_gemm(st, ws+O_IAKQ_IM, ws+O_MOD, ws+O_TP+9, 64,9,32, 32,9,18, 2048,0,1.f, 1,0,0, 1,288,0, 1,1152,0);
  launch_gemm(st, ws+O_TP, ws+O_WCSKQ, ws+O_Q,       64,64,18, 18,64,64, 2048,0,1.f, 1,1152,0, 1,0,0, 1,4096,0);
  axpb_add<<<CDIV(2048*4096,256),256,0,st>>>(ws+O_Q, ws+O_R, q_sw, q_sb, 256, 8, 4096);

  // ---- V
  mode_mul_expand<<<CDIV(256*8*16*9,256),256,0,st>>>(ws+O_MKQV, v_spec, ws+O_MOD, 256, 8, 16, 9);
  launch_gemm(st, ws+O_IAV_RE, ws+O_MOD, ws+O_TP,   128,9,32, 32,9,18, 2048,0,1.f, 1,0,0, 1,288,0, 1,2304,0);
  launch_gemm(st, ws+O_IAV_IM, ws+O_MOD, ws+O_TP+9, 128,9,32, 32,9,18, 2048,0,1.f, 1,0,0, 1,288,0, 1,2304,0);
  launch_gemm(st, ws+O_TP, ws+O_WCSV, ws+O_V,       128,128,18, 18,128,128, 2048,0,1.f, 1,2304,0, 1,0,0, 1,16384,0);
  axpb_add<<<CDIV(2048*16384,256),256,0,st>>>(ws+O_V, ws+O_TN, v_sw, v_sb, 256, 8, 16384);

  // ---- attention (TDM-staged GEMMs)
  launch_gemm_tdm(st, ws+O_Q, ws+O_K, ws+O_SC, 64,64,4096, 32768,32768,64, 32,1, 1.f/64.f,
                  8,2097152,4096, 8,2097152,4096, 1,4096,0);
  softmax64<<<2048,64,0,st>>>(ws+O_SC);
  launch_gemm_tdm(st, ws+O_SC, ws+O_V, ws+O_AO, 64,16384,64, 64,131072,131072, 32,0,1.f,
                  1,4096,0, 8,8388608,16384, 8,8388608,16384);

  // ---- projection FNO layer (heads -> 1), modes 32x17
  launch_gemm(st, ws+O_AO, ws+O_WC17, ws+O_SA,      128,17,128, 128,17,17, 2048,0,1.f, 1,16384,0, 1,0,0, 1,4352,0);
  launch_gemm(st, ws+O_AO, ws+O_WS17, ws+O_SA+2176, 128,17,128, 128,17,17, 2048,0,1.f, 1,16384,0, 1,0,0, 1,4352,0);
  launch_gemm(st, ws+O_FH16, ws+O_SA, ws+O_MOD,     64,17,256,  256,17,17, 2048,0,1.f, 1,0,0, 1,4352,0, 1,1088,0);
  mode_mul_reduce<<<CDIV(256*32*17,256),256,0,st>>>(ws+O_MOD, p_spec, ws+O_SMOD, 256, 8, 32, 17);
  launch_gemm(st, ws+O_IAM_RE, ws+O_SMOD, ws+O_TP,    128,17,64, 64,17,34, 256,0,1.f, 1,0,0, 1,1088,0, 1,4352,0);
  launch_gemm(st, ws+O_IAM_IM, ws+O_SMOD, ws+O_TP+17, 128,17,64, 64,17,34, 256,0,1.f, 1,0,0, 1,1088,0, 1,4352,0);
  launch_gemm(st, ws+O_TP, ws+O_WCSM, ws+O_T1,        128,128,34, 34,128,128, 256,0,1.f, 1,4352,0, 1,0,0, 1,16384,0);
  proj_skip_add<<<CDIV(256*16384,256),256,0,st>>>(ws+O_T1, ws+O_AO, p_sw, p_sb, 256, 8, 16384);

  // ---- a = inorm(proj + x); an = inorm(a)
  inorm_kernel<<<256,256,0,st>>>(ws+O_T1, xin, nullptr, ws+O_A, 16384, an_g, an_b);
  inorm_kernel<<<256,256,0,st>>>(ws+O_A, nullptr, nullptr, ws+O_AN, 16384, n2_g, n2_b);

  // ---- mixer layer 1
  launch_gemm(st, ws+O_AN, ws+O_WC17, ws+O_SA,      128,17,128, 128,17,17, 256,0,1.f, 1,16384,0, 1,0,0, 1,4352,0);
  launch_gemm(st, ws+O_AN, ws+O_WS17, ws+O_SA+2176, 128,17,128, 128,17,17, 256,0,1.f, 1,16384,0, 1,0,0, 1,4352,0);
  launch_gemm(st, ws+O_FH16, ws+O_SA, ws+O_MOD,     64,17,256,  256,17,17, 256,0,1.f, 1,0,0, 1,4352,0, 1,1088,0);
  mode_mul_expand<<<CDIV(256*32*17,256),256,0,st>>>(ws+O_MOD, m1_spec, ws+O_SMOD, 256, 1, 32, 17);
  launch_gemm(st, ws+O_IAM_RE, ws+O_SMOD, ws+O_TP,    128,17,64, 64,17,34, 256,0,1.f, 1,0,0, 1,1088,0, 1,4352,0);
  launch_gemm(st, ws+O_IAM_IM, ws+O_SMOD, ws+O_TP+17, 128,17,64, 64,17,34, 256,0,1.f, 1,0,0, 1,1088,0, 1,4352,0);
  launch_gemm(st, ws+O_TP, ws+O_WCSM, ws+O_T2,        128,128,34, 34,128,128, 256,0,1.f, 1,4352,0, 1,0,0, 1,16384,0);
  inorm_kernel<<<256,256,0,st>>>(ws+O_T2, nullptr, nullptr, ws+O_T2, 16384, nullptr, nullptr);
  mixer_combine<<<CDIV(4194304,256),256,0,st>>>(ws+O_H, ws+O_T2, ws+O_AN, m1_sw, m1_sb, 4194304LL, 1);

  // ---- mixer layer 2
  launch_gemm(st, ws+O_H, ws+O_WC17, ws+O_SA,      128,17,128, 128,17,17, 256,0,1.f, 1,16384,0, 1,0,0, 1,4352,0);
  launch_gemm(st, ws+O_H, ws+O_WS17, ws+O_SA+2176, 128,17,128, 128,17,17, 256,0,1.f, 1,16384,0, 1,0,0, 1,4352,0);
  launch_gemm(st, ws+O_FH16, ws+O_SA, ws+O_MOD,    64,17,256,  256,17,17, 256,0,1.f, 1,0,0, 1,4352,0, 1,1088,0);
  mode_mul_expand<<<CDIV(256*32*17,256),256,0,st>>>(ws+O_MOD, m2_spec, ws+O_SMOD, 256, 1, 32, 17);
  launch_gemm(st, ws+O_IAM_RE, ws+O_SMOD, ws+O_TP,    128,17,64, 64,17,34, 256,0,1.f, 1,0,0, 1,1088,0, 1,4352,0);
  launch_gemm(st, ws+O_IAM_IM, ws+O_SMOD, ws+O_TP+17, 128,17,64, 64,17,34, 256,0,1.f, 1,0,0, 1,1088,0, 1,4352,0);
  launch_gemm(st, ws+O_TP, ws+O_WCSM, ws+O_T2,        128,128,34, 34,128,128, 256,0,1.f, 1,4352,0, 1,0,0, 1,16384,0);
  inorm_kernel<<<256,256,0,st>>>(ws+O_T2, nullptr, nullptr, ws+O_T2, 16384, nullptr, nullptr);
  mixer_combine<<<CDIV(4194304,256),256,0,st>>>(ws+O_T2, ws+O_T2, ws+O_H, m2_sw, m2_sb, 4194304LL, 0);

  // ---- output
  inorm_kernel<<<256,256,0,st>>>(ws+O_T2, nullptr, ws+O_A, out, 16384, mo_g, mo_b);
}
